// AttentionLayer_23785528885493
// MI455X (gfx1250) — matrix operand order repaired and checked by reading, not compiled, not run on hardware
//
#include <hip/hip_runtime.h>
#include <hip/hip_bf16.h>

// MI455X attention:  out = softmax((xWq^T+bq)(xWk^T+bk)^T / sqrt(H)) (xWv^T+bv)
// B=4, S=2048, H=1024.  Bandwidth-bound (~180MB @ 23.3TB/s).
// Pipeline: (0) one-pass f32->f16 conversion of x and W, (1) QKV WMMA GEMM
// (V stored transposed [B,H,S]), (2) per-16-query flash block: scores in VGPR
// accumulators, softmax via shfl_xor + LDS, P in 64KB LDS, O = P@V.
// Hot loops: peeled first step uses inline-0 C (no accumulator-init movs);
// steady state is branch-free ping-pong double buffering (no loop-carried
// register copies); speculative tail loads land in a workspace/LDS pad.

typedef __attribute__((ext_vector_type(16))) _Float16 v16h;
typedef __attribute__((ext_vector_type(8)))  _Float16 v8h;
typedef __attribute__((ext_vector_type(8)))  float    v8f;

constexpr int Hd = 1024;   // model dim
constexpr int Sd = 2048;   // sequence
constexpr int Bn = 4;      // batch

#define WMMA16(a, b, c) __builtin_amdgcn_wmma_f32_16x16x32_f16( \
    false, (a), false, (b), (short)0, (c), false, false)

// ---- fragment loader (WMMA 16x16x32 f16 A operand, contiguous k order; B matches) ----
// lane l: row = l&15 ; K = {kh..kh+15}, kh = (l>>4)*16
static __device__ __forceinline__ v16h ld_a16(const _Float16* p) {
  v8h lo = *(const v8h*)(p);
  v8h hi = *(const v8h*)(p + 8);
  v16h r;
#pragma unroll
  for (int i = 0; i < 8; ++i) { r[i] = lo[i]; r[i + 8] = hi[i]; }
  return r;
}

// ------------------------------------------------------------------
// Kernel 0: f32 -> f16 conversion, 8 elements/thread, one pass.
// blockIdx.y selects buffer: 0=x, 1=Wq, 2=Wk, 3=Wv.
// ------------------------------------------------------------------
__global__ __launch_bounds__(256)
void cvt_kernel(const float* __restrict__ s0, _Float16* __restrict__ d0, int c0,
                const float* __restrict__ s1, _Float16* __restrict__ d1, int c1,
                const float* __restrict__ s2, _Float16* __restrict__ d2, int c2,
                const float* __restrict__ s3, _Float16* __restrict__ d3, int c3) {
  const int sel = blockIdx.y;
  const float* s = (sel == 0) ? s0 : (sel == 1) ? s1 : (sel == 2) ? s2 : s3;
  _Float16*   d = (sel == 0) ? d0 : (sel == 1) ? d1 : (sel == 2) ? d2 : d3;
  const int   n = (sel == 0) ? c0 : (sel == 1) ? c1 : (sel == 2) ? c2 : c3;
  const int i = (blockIdx.x * 256 + threadIdx.x) * 8;
  if (i >= n) return;
  v8f v = *(const v8f*)(s + i);
  v8h h;
#pragma unroll
  for (int e = 0; e < 8; ++e) h[e] = (_Float16)v[e];
  *(v8h*)(d + i) = h;
}

// ------------------------------------------------------------------
// Kernel 1: fused QKV projection.  y = x @ W^T + b  (torch Linear).
// A[M=s,K=h] = X16[s,h] ; B[K=h,N=o] = W16[o,h] (rows contiguous in h).
// z==0 -> Q16 row-major, z==1 -> K16 row-major, z==2 -> Vt16 [B,H,S].
// Ping-pong pipelined k-loop, inline-0 C on step 0.
// ------------------------------------------------------------------
__global__ __launch_bounds__(256)
void qkv_kernel(const _Float16* __restrict__ X16,
                const _Float16* __restrict__ W16,   // [3][Hd][Hd] (+pad)
                const float* __restrict__ bq, const float* __restrict__ bk,
                const float* __restrict__ bv,
                _Float16* __restrict__ Q16, _Float16* __restrict__ K16,
                _Float16* __restrict__ Vt16) {
  const int lane = threadIdx.x & 31;
  const int wave = threadIdx.x >> 5;
  const int z = blockIdx.z;
  const _Float16* W = W16 + (size_t)z * Hd * Hd;
  const float* bias = (z == 0) ? bq : (z == 1) ? bk : bv;

  const int m0 = blockIdx.x * 128 + wave * 16;  // global row (b*S + s)
  const int n0 = blockIdx.y * 64;               // output feature

  const int ar  = lane & 15;          // A row within tile
  const int akh = (lane >> 4) * 16;   // A K sub-offset
  const int bn  = lane & 15;          // B column (N)
  const int bkg = (lane >> 4) * 16;   // B K group
  const int mhi = (lane >> 4) * 8;    // D row offset

  const _Float16* arow = X16 + (m0 + ar) * Hd + akh;
  const _Float16* bcol = W + (n0 + bn) * Hd + bkg;  // + j*16*Hd + k0

  v8f acc[4];
  v16h a0 = ld_a16(arow);
  v16h a1 = ld_a16(arow + 32);
  v16h b0[4], b1[4];
#pragma unroll
  for (int j = 0; j < 4; ++j) b0[j] = *(const v16h*)(bcol + j * 16 * Hd);
#pragma unroll
  for (int j = 0; j < 4; ++j) b1[j] = *(const v16h*)(bcol + j * 16 * Hd + 32);

  {                                   // step 0: C = inline 0
    v8f z8 = {};
#pragma unroll
    for (int j = 0; j < 4; ++j) acc[j] = WMMA16(a0, b0[j], z8);
  }
  a0 = ld_a16(arow + 64);
#pragma unroll
  for (int j = 0; j < 4; ++j) b0[j] = *(const v16h*)(bcol + j * 16 * Hd + 64);
#pragma unroll
  for (int j = 0; j < 4; ++j) acc[j] = WMMA16(a1, b1[j], acc[j]);   // step 1
  a1 = ld_a16(arow + 96);
#pragma unroll
  for (int j = 0; j < 4; ++j) b1[j] = *(const v16h*)(bcol + j * 16 * Hd + 96);

  for (int k0 = 64; k0 < Hd - 32; k0 += 64) {   // steps 2..31, branch-free
#pragma unroll
    for (int j = 0; j < 4; ++j) acc[j] = WMMA16(a0, b0[j], acc[j]); // k0
    a0 = ld_a16(arow + k0 + 64);                // speculative at tail (pad)
#pragma unroll
    for (int j = 0; j < 4; ++j) b0[j] = *(const v16h*)(bcol + j * 16 * Hd + k0 + 64);
    __builtin_prefetch((const void*)(arow + k0 + 128), 0, 3);
#pragma unroll
    for (int j = 0; j < 4; ++j) acc[j] = WMMA16(a1, b1[j], acc[j]); // k0+32
    a1 = ld_a16(arow + k0 + 96);
#pragma unroll
    for (int j = 0; j < 4; ++j) b1[j] = *(const v16h*)(bcol + j * 16 * Hd + k0 + 96);
  }

  if (z < 2) {
    _Float16* out = (z == 0) ? Q16 : K16;
#pragma unroll
    for (int j = 0; j < 4; ++j) {
      const int n = n0 + j * 16 + bn;
      const float bb = bias[n];
#pragma unroll
      for (int e = 0; e < 8; ++e) {
        const int row = m0 + mhi + e;
        out[row * Hd + n] = (_Float16)(acc[j][e] + bb);
      }
    }
  } else {
    // transposed store: Vt[b][n][s]; e=0..7 are consecutive s -> 16B store.
    const int b = m0 / Sd;            // blocks never straddle a batch
    const int s0 = (m0 - b * Sd) + mhi;
#pragma unroll
    for (int j = 0; j < 4; ++j) {
      const int n = n0 + j * 16 + bn;
      const float bb = bias[n];
      v8h hv;
#pragma unroll
      for (int e = 0; e < 8; ++e) hv[e] = (_Float16)(acc[j][e] + bb);
      *(v8h*)(Vt16 + ((size_t)b * Hd + n) * Sd + s0) = hv;
    }
  }
}

// ------------------------------------------------------------------
// Kernel 2: attention for one (batch b, 16-query block m0).
// Phase 1: wave w -> raw scores for keys [w*256, +256) in 16 WMMA
//          accumulators; softmax stats via shfl_xor + LDS; unnormalized
//          exp() written as f16 to LDS P (16 x 2048).
// Phase 2: wave w -> O[:, w*128 .. +128) = P @ V; 1/rowsum in epilogue.
// Both phases: ping-pong A pipelining, pipelined B inside the tile loop.
// ------------------------------------------------------------------
__global__ __launch_bounds__(256)
void attn_kernel(const _Float16* __restrict__ Q16, const _Float16* __restrict__ K16,
                 const _Float16* __restrict__ Vt16, float* __restrict__ out) {
  extern __shared__ char smem[];
  _Float16* Pb  = (_Float16*)smem;                    // 16 * 2048 f16 = 64KB
  float* redA   = (float*)(smem + 16 * Sd * 2);       // 8*16 max partials
  float* redB   = redA + 128;                         // 8*16 sum partials
  float* linv   = redB + 128;                         // 16 inverse row sums

  const int lane = threadIdx.x & 31;
  const int wave = threadIdx.x >> 5;
  const int b  = blockIdx.y;
  const int m0 = blockIdx.x * 16;

  const int ar  = lane & 15;
  const int akh = (lane >> 4) * 16;
  const int bn  = lane & 15;
  const int bkg = (lane >> 4) * 16;
  const int mhi = (lane >> 4) * 8;

  const float scale = 0.03125f;               // 1/sqrt(1024)
  const float l2e   = 1.4426950408889634f;

  // ---------- phase 1: scores ----------
  v8f sc[16];
  const _Float16* qrow = Q16 + ((size_t)(b * Sd + m0 + ar)) * Hd + akh;
  const int key0 = wave * 256;
  const _Float16* krow = K16 + ((size_t)b * Sd + key0 + bn) * Hd + bkg;

  // one k-step of the score GEMM (16 key tiles, pipelined B fragments)
  auto kstep = [&](const v16h& av, int k0, bool first) {
    v16h bf = *(const v16h*)(krow + k0);
#pragma unroll
    for (int t = 0; t < 16; ++t) {
      v16h bf_n;
      if (t < 15) bf_n = *(const v16h*)(krow + (t + 1) * 16 * Hd + k0);
      if (first) { v8f z8 = {}; sc[t] = WMMA16(av, bf, z8); }
      else       { sc[t] = WMMA16(av, bf, sc[t]); }
      bf = bf_n;
    }
  };

  v16h a0 = ld_a16(qrow);
  v16h a1 = ld_a16(qrow + 32);
  kstep(a0, 0, true);
  a0 = ld_a16(qrow + 64);
  kstep(a1, 32, false);
  a1 = ld_a16(qrow + 96);
  for (int k0 = 64; k0 < Hd - 32; k0 += 64) {   // steps 2..31
    kstep(a0, k0, false);
    a0 = ld_a16(qrow + k0 + 64);                // speculative at tail
    __builtin_prefetch((const void*)(krow + k0 + 64), 0, 3);
    kstep(a1, k0 + 32, false);
    a1 = ld_a16(qrow + k0 + 96);
  }

  // row max over this wave's 256 keys (scaled)
  float vmax[8];
#pragma unroll
  for (int e = 0; e < 8; ++e) {
    float m = -1e30f;
#pragma unroll
    for (int t = 0; t < 16; ++t) m = fmaxf(m, sc[t][e]);
    vmax[e] = m * scale;
  }
#pragma unroll
  for (int off = 1; off < 16; off <<= 1)
#pragma unroll
    for (int e = 0; e < 8; ++e)
      vmax[e] = fmaxf(vmax[e], __shfl_xor(vmax[e], off, 32));
  if ((lane & 15) == 0) {
#pragma unroll
    for (int e = 0; e < 8; ++e) redA[wave * 16 + mhi + e] = vmax[e];
  }
  __syncthreads();
  float rmax[8];
#pragma unroll
  for (int e = 0; e < 8; ++e) {
    float m = -1e30f;
#pragma unroll
    for (int w = 0; w < 8; ++w) m = fmaxf(m, redA[w * 16 + mhi + e]);
    rmax[e] = m;
  }

  // exp, partial row sums, write unnormalized P to LDS
  float vsum[8] = {};
#pragma unroll
  for (int t = 0; t < 16; ++t) {
#pragma unroll
    for (int e = 0; e < 8; ++e) {
      float p = exp2f((sc[t][e] * scale - rmax[e]) * l2e);
      vsum[e] += p;
      Pb[(mhi + e) * Sd + key0 + t * 16 + bn] = (_Float16)p;
    }
  }
#pragma unroll
  for (int off = 1; off < 16; off <<= 1)
#pragma unroll
    for (int e = 0; e < 8; ++e) vsum[e] += __shfl_xor(vsum[e], off, 32);
  if ((lane & 15) == 0) {
#pragma unroll
    for (int e = 0; e < 8; ++e) redB[wave * 16 + mhi + e] = vsum[e];
  }
  __syncthreads();
  if (wave == 0 && (lane == 0 || lane == 16)) {
#pragma unroll
    for (int e = 0; e < 8; ++e) {
      float s = 0.f;
#pragma unroll
      for (int w = 0; w < 8; ++w) s += redB[w * 16 + mhi + e];
      linv[mhi + e] = 1.0f / s;
    }
  }
  __syncthreads();

  // ---------- phase 2: O = P @ V ----------
  v8f oacc[8];
  const int n0 = wave * 128;
  const _Float16* vrow = Vt16 + ((size_t)b * Hd + n0 + bn) * Sd + bkg;
  const _Float16* prow = Pb + ar * Sd + akh;

  auto vstep = [&](const v16h& av, int k0, bool first) {
    v16h bf = *(const v16h*)(vrow + k0);
#pragma unroll
    for (int j = 0; j < 8; ++j) {
      v16h bf_n;
      if (j < 7) bf_n = *(const v16h*)(vrow + (j + 1) * 16 * Sd + k0);
      if (first) { v8f z8 = {}; oacc[j] = WMMA16(av, bf, z8); }
      else       { oacc[j] = WMMA16(av, bf, oacc[j]); }
      bf = bf_n;
    }
  };

  v16h p0 = ld_a16(prow);                      // ds_load_b128 x2
  v16h p1 = ld_a16(prow + 32);
  vstep(p0, 0, true);
  p0 = ld_a16(prow + 64);
  vstep(p1, 32, false);
  p1 = ld_a16(prow + 96);
  for (int k0 = 64; k0 < Sd - 32; k0 += 64) {  // steps 2..63
    vstep(p0, k0, false);
    p0 = ld_a16(prow + k0 + 64);               // speculative at tail (LDS slack)
    __builtin_prefetch((const void*)(vrow + k0 + 64), 0, 3);
    vstep(p1, k0 + 32, false);
    p1 = ld_a16(prow + k0 + 96);
  }

#pragma unroll
  for (int j = 0; j < 8; ++j) {
    const int n = n0 + j * 16 + bn;
#pragma unroll
    for (int e = 0; e < 8; ++e) {
      const int q = mhi + e;
      out[((size_t)(b * Sd + m0 + q)) * Hd + n] = oacc[j][e] * linv[q];
    }
  }
}

// ------------------------------------------------------------------
extern "C" void kernel_launch(void* const* d_in, const int* in_sizes, int n_in,
                              void* d_out, int out_size, void* d_ws, size_t ws_size,
                              hipStream_t stream) {
  (void)in_sizes; (void)n_in; (void)out_size; (void)ws_size;
  const float* x  = (const float*)d_in[0];
  const float* Wq = (const float*)d_in[1];
  const float* bq = (const float*)d_in[2];
  const float* Wk = (const float*)d_in[3];
  const float* bk = (const float*)d_in[4];
  const float* Wv = (const float*)d_in[5];
  const float* bv = (const float*)d_in[6];
  float* out = (float*)d_out;

  const size_t nBSH = (size_t)Bn * Sd * Hd;   // 8.4M elements
  const size_t nHH  = (size_t)Hd * Hd;        // 1.05M elements

  _Float16* Q16  = (_Float16*)d_ws;           // 16 MB
  _Float16* K16  = Q16 + nBSH;                // 16 MB
  _Float16* Vt16 = K16 + nBSH;                // 16 MB (transposed [B,H,S])
  _Float16* X16  = Vt16 + nBSH;               // 16 MB
  _Float16* W16  = X16 + nBSH;                // 6 MB ([3][Hd][Hd]); 64KB pad after

  // 0) one-pass f32 -> f16 of activations + weights
  dim3 g0((unsigned)(nBSH / (256 * 8)), 4);
  cvt_kernel<<<g0, 256, 0, stream>>>(x, X16, (int)nBSH,
                                     Wq, W16 + 0 * nHH, (int)nHH,
                                     Wk, W16 + 1 * nHH, (int)nHH,
                                     Wv, W16 + 2 * nHH, (int)nHH);

  // 1) QKV projections
  dim3 g1(Bn * Sd / 128, Hd / 64, 3);
  qkv_kernel<<<g1, 256, 0, stream>>>(X16, W16, bq, bk, bv, Q16, K16, Vt16);

  // 2) attention (dynamic LDS: 64KB P + reduction scratch + 256B slack)
  size_t sm = (size_t)16 * Sd * sizeof(_Float16) + (2 * 128 + 16) * sizeof(float) + 256;
  dim3 g2(Sd / 16, Bn);
  attn_kernel<<<g2, 256, sm, stream>>>(Q16, K16, Vt16, out);
}
